// Classifier_3788161155197
// MI455X (gfx1250) — matrix operand order repaired and checked by reading, not compiled, not run on hardware
//
#include <hip/hip_runtime.h>
#include <math.h>

// ---------------------------------------------------------------------------
// Problem dims (from reference): N=4096 sentences, E=768, H=512
// ---------------------------------------------------------------------------
#define NSENT 4096
#define EDIM  768
#define HDIM  512
#define G4H   2048   // 4*H
#define D2H   1024   // 2*H

typedef __bf16 bf16;
typedef bf16  v16bf __attribute__((ext_vector_type(16)));
typedef bf16  v8bf  __attribute__((ext_vector_type(8)));
typedef float v8f   __attribute__((ext_vector_type(8)));

__device__ __forceinline__ float sigf(float x) { return 1.0f / (1.0f + expf(-x)); }

// ---------------------------------------------------------------------------
// f32 -> bf16 convert
// ---------------------------------------------------------------------------
__global__ void cvt_bf16_kernel(const float* __restrict__ in, bf16* __restrict__ out, int n) {
    int i = blockIdx.x * 256 + threadIdx.x;
    if (i < n) out[i] = (bf16)in[i];
}

// ---------------------------------------------------------------------------
// WMMA GEMM:  C[M,N] = epi( A[M,K] @ B[N,K]^T + bias[N] )
// A,B bf16 row-major (K contiguous), accum f32 via v_wmma_f32_16x16x32_bf16.
// Block = 256 threads = 8 waves; wave computes a 16x64 strip (4 C tiles).
// Grid = (N/64, M/128). K is a multiple of 64 at every call site.
// K-loop is software-pipelined with double-buffered fragments so the
// scheduler can overlap VMEM latency with the matrix pipe (avoid the
// s_wait_loadcnt 0x0 drain before every WMMA seen in the naive version).
// Fragment packing follows CDNA5 ISA 7.12.2 16-bit layouts.
// ---------------------------------------------------------------------------
enum { EPI_RAW = 0, EPI_TANH_F32 = 1, EPI_TANH_BF16 = 2 };

template <int EPI>
__global__ __launch_bounds__(256)
void wmma_gemm_kernel(const bf16* __restrict__ A, const bf16* __restrict__ B,
                      const float* __restrict__ bias,
                      float* __restrict__ Cf, bf16* __restrict__ Cb,
                      int M, int N, int K) {
    const int lane  = threadIdx.x & 31;
    const int wave  = threadIdx.x >> 5;
    const int n0    = blockIdx.x * 64;
    const int m0    = blockIdx.y * 128 + wave * 16;
    const int mrow  = m0 + (lane & 15);
    const int khalf = lane >> 4;

    const bf16* Arow = A + (size_t)mrow * K;
    const bf16* Brow[4];
#pragma unroll
    for (int c = 0; c < 4; ++c)
        Brow[c] = B + (size_t)(n0 + c * 16 + (lane & 15)) * K;

    // A fragment: lanes 0-15 hold K {0..7,16..23}, lanes 16-31 {8..15,24..31}
    auto loadA = [&](int k0, v16bf& f) {
        v8bf lo = *(const v8bf*)(Arow + k0 + khalf * 8);
        v8bf hi = *(const v8bf*)(Arow + k0 + 16 + khalf * 8);
#pragma unroll
        for (int e = 0; e < 8; ++e) { f[e] = lo[e]; f[e + 8] = hi[e]; }
    };
    // B fragment: lane%16 = column, lanes 0-15 K {0..7,16..23}, lanes 16-31 {8..15,24..31}
    auto loadB4 = [&](int k0, v16bf* f) {
#pragma unroll
        for (int c = 0; c < 4; ++c) {
            const bf16* p = Brow[c] + k0 + khalf * 8;
            v8bf lo = *(const v8bf*)(p);
            v8bf hi = *(const v8bf*)(p + 16);
#pragma unroll
            for (int e = 0; e < 8; ++e) { f[c][e] = lo[e]; f[c][e + 8] = hi[e]; }
        }
    };

    v8f acc[4] = {};
    auto wmma4 = [&](const v16bf& a, const v16bf* b) {
#pragma unroll
        for (int c = 0; c < 4; ++c)
            acc[c] = __builtin_amdgcn_wmma_f32_16x16x32_bf16(
                false, a, false, b[c], (short)0, acc[c], false, false);
    };

    // Double-buffered, 2x-unrolled pipeline over K (K % 64 == 0)
    v16bf a0, a1, b0[4], b1[4];
    loadA(0, a0);  loadB4(0, b0);
    loadA(32, a1); loadB4(32, b1);
    int k0 = 0;
    for (; k0 < K - 64; k0 += 64) {
        wmma4(a0, b0);                             // consume k0
        loadA(k0 + 64, a0); loadB4(k0 + 64, b0);   // prefetch k0+64
        wmma4(a1, b1);                             // consume k0+32
        loadA(k0 + 96, a1); loadB4(k0 + 96, b1);   // prefetch k0+96
    }
    wmma4(a0, b0);                                 // k0 == K-64
    wmma4(a1, b1);                                 // k0 == K-32

    // Epilogue: C layout -> lane%16 = col, rows m0 + (lane>>4)*8 + r
    const int rbase = m0 + (lane >> 4) * 8;
#pragma unroll
    for (int c = 0; c < 4; ++c) {
        const int ncol = n0 + c * 16 + (lane & 15);
        const float bv = bias[ncol];
#pragma unroll
        for (int r = 0; r < 8; ++r) {
            float v = acc[c][r] + bv;
            size_t off = (size_t)(rbase + r) * N + ncol;
            if (EPI == EPI_RAW)           Cf[off] = v;
            else if (EPI == EPI_TANH_F32) Cf[off] = tanhf(v);
            else                          Cb[off] = (bf16)tanhf(v);
        }
    }
}

// ---------------------------------------------------------------------------
// Context-encoder LSTM cell epilogue (h = c = 0):
//   h = sigmoid(o) * tanh( sigmoid(i) * tanh(g) )
// gates: [4096, 2048] f32 (i|f|g|o). Writes outp2 f32 + bf16 at column base.
// ---------------------------------------------------------------------------
__global__ void hgate_kernel(const float* __restrict__ gates,
                             float* __restrict__ o2f, bf16* __restrict__ o2b,
                             int colbase) {
    int i = blockIdx.x * 256 + threadIdx.x;
    if (i >= NSENT * HDIM) return;
    int m = i >> 9, u = i & 511;
    const float* gr = gates + (size_t)m * G4H;
    float gi = gr[u], gg = gr[1024 + u], go = gr[1536 + u];
    float c2 = sigf(gi) * tanhf(gg);                 // sigmoid(f)*c vanishes (c==0)
    float h  = sigf(go) * tanhf(c2);
    size_t off = (size_t)m * D2H + colbase + u;
    o2f[off] = h;
    o2b[off] = (bf16)h;
}

// s[m] = att[m,:] . ws2_w + ws2_b
__global__ void smv_kernel(const float* __restrict__ att, const float* __restrict__ w2,
                           const float* __restrict__ b2, float* __restrict__ s) {
    int m = blockIdx.x * 256 + threadIdx.x;
    if (m >= NSENT) return;
    const float* r = att + (size_t)m * D2H;
    float acc = b2[0];
    for (int k = 0; k < D2H; ++k) acc += r[k] * w2[k];
    s[m] = acc;
}

// scale = sum_j softmax(s)_j   (single block; faithful to reference ordering-ish)
__global__ void scale_kernel(const float* __restrict__ s, float* __restrict__ scale) {
    __shared__ float red[1024];
    int tid = threadIdx.x;
    float mx = -3.4e38f;
    for (int i = tid; i < NSENT; i += 1024) mx = fmaxf(mx, s[i]);
    red[tid] = mx; __syncthreads();
    for (int st = 512; st > 0; st >>= 1) { if (tid < st) red[tid] = fmaxf(red[tid], red[tid + st]); __syncthreads(); }
    mx = red[0]; __syncthreads();
    float sm = 0.f;
    for (int i = tid; i < NSENT; i += 1024) sm += expf(s[i] - mx);
    red[tid] = sm; __syncthreads();
    for (int st = 512; st > 0; st >>= 1) { if (tid < st) red[tid] += red[tid + st]; __syncthreads(); }
    float S = red[0]; __syncthreads();
    float q = 0.f;
    for (int i = tid; i < NSENT; i += 1024) q += expf(s[i] - mx) / S;
    red[tid] = q; __syncthreads();
    for (int st = 512; st > 0; st >>= 1) { if (tid < st) red[tid] += red[tid + st]; __syncthreads(); }
    if (tid == 0) scale[0] = red[0];
}

__global__ void scale_sent_kernel(const float* __restrict__ o2f,
                                  const float* __restrict__ scale,
                                  bf16* __restrict__ sout) {
    int i = blockIdx.x * 256 + threadIdx.x;
    if (i < NSENT * D2H) sout[i] = (bf16)(o2f[i] * scale[0]);
}

__global__ void rec_init_kernel(float* __restrict__ hbuf, unsigned int* __restrict__ ctr) {
    int i = blockIdx.x * 256 + threadIdx.x;
    if (i < 2 * 2 * HDIM) hbuf[i] = 0.0f;
    if (i == 0) *ctr = 0u;
}

// ---------------------------------------------------------------------------
// Persistent bidirectional LSTM scan. 32 workgroups (16 fwd, 16 bwd), 512 thr.
// Each WG owns 128 rows of W_hh (4 gates x 32 hidden units) in LDS (f32,
// padded to 132 dwords per 128-k quarter => conflict-free b128 reads).
// Per step: 128 length-512 dots from LDS, cell update on 32 unit-threads,
// h exchanged via L2 ping-pong buffer, steps fenced by a device-scope
// atomic arrive counter with s_sleep spin. X = precomputed x@W_ih^T + b.
// LDS: 128*528*4 + (512+512+32)*4 = 274560 B (CDNA5: 320 KB/WGP).
// ---------------------------------------------------------------------------
#define REC_LDS_BYTES (128 * 528 * 4 + (512 + 512 + 32) * 4)

__global__ __launch_bounds__(512, 1)
void lstm_rec_kernel(const float* __restrict__ Xf, const float* __restrict__ Xb,
                     const float* __restrict__ Whf, const float* __restrict__ Whb,
                     float* __restrict__ hbuf, unsigned int* __restrict__ ctr,
                     float* __restrict__ ipred) {
    extern __shared__ char smem[];
    float* Wl   = (float*)smem;            // [128][4 quarters][132]
    float* hsh  = Wl + 128 * 528;          // [512]
    float* part = hsh + 512;               // [512]
    float* csh  = part + 512;              // [32]

    const int b   = blockIdx.x;
    const int dir = b >> 4;                // 0 = forward, 1 = backward
    const int sl  = b & 15;                // slice: hidden units [sl*32, sl*32+32)
    const float* W = dir ? Whb : Whf;      // [2048,512] row-major, rows = gate*512+unit
    const float* X = dir ? Xb : Xf;        // [4096,2048] precomputed input gates
    const int tid = threadIdx.x;

    // Stage this WG's 128 W_hh rows into padded LDS
    for (int idx = tid; idx < 128 * 512; idx += 512) {
        int r = idx >> 9, kg = idx & 511;
        int q = r >> 5, u = r & 31;
        Wl[r * 528 + (kg >> 7) * 132 + (kg & 127)] = W[(size_t)(q * 512 + sl * 32 + u) * 512 + kg];
    }
    if (tid < 32) csh[tid] = 0.0f;
    __syncthreads();

    float* hb = hbuf + dir * (2 * HDIM);   // ping-pong [2][512]

    for (int t = 0; t < NSENT; ++t) {
        // fetch h_{t-1}
        hsh[tid] = hb[(t & 1) * HDIM + tid];
        // prefetch next step's input-gate row while we compute
        if (tid < 32) {
            int nrow = dir ? (NSENT - 2 - t) : (t + 1);
            if (nrow >= 0 && nrow < NSENT)
                __builtin_prefetch(X + (size_t)nrow * G4H + sl * 32 + tid, 0, 1);
        }
        __syncthreads();

        // 128 dot-products, 4 threads per output row (one k-quarter each)
        const int r  = tid >> 2;
        const int q4 = tid & 3;
        const float* wrow = Wl + r * 528 + q4 * 132;
        const float* hp   = hsh + q4 * 128;
        float acc = 0.0f;
#pragma unroll 8
        for (int k = 0; k < 128; k += 4) {
            float4 w4 = *(const float4*)(wrow + k);
            float4 h4 = *(const float4*)(hp + k);
            acc += w4.x * h4.x + w4.y * h4.y + w4.z * h4.z + w4.w * h4.w;
        }
        part[tid] = acc;
        __syncthreads();

        if (tid < 32) {
            const int u   = tid;
            const int row = dir ? (NSENT - 1 - t) : t;
            const float* xr = X + (size_t)row * G4H + sl * 32 + u;
#define GSUM(q) (part[((q)*32 + u) * 4 + 0] + part[((q)*32 + u) * 4 + 1] + \
                 part[((q)*32 + u) * 4 + 2] + part[((q)*32 + u) * 4 + 3])
            float gi = GSUM(0) + xr[0];
            float gf = GSUM(1) + xr[512];
            float gg = GSUM(2) + xr[1024];
            float go = GSUM(3) + xr[1536];
#undef GSUM
            float c  = csh[u];
            float c2 = sigf(gf) * c + sigf(gi) * tanhf(gg);
            float h2 = sigf(go) * tanhf(c2);
            csh[u] = c2;
            hb[((t + 1) & 1) * HDIM + sl * 32 + u] = h2;
            ipred[(size_t)row * D2H + dir * HDIM + sl * 32 + u] = h2;
            __threadfence();
        }
        __syncthreads();

        // device-scope arrive counter: all 32 WGs lockstep per timestep
        if (tid == 0) {
            atomicAdd(ctr, 1u);
            const unsigned int target = 32u * (unsigned int)(t + 1);
            while (__hip_atomic_load(ctr, __ATOMIC_RELAXED, __HIP_MEMORY_SCOPE_AGENT) < target)
                __builtin_amdgcn_s_sleep(1);
        }
        __syncthreads();
        __threadfence();
    }
}

// ---------------------------------------------------------------------------
// Window max-pool + concat(disc_feat) + final 2-way linear. One block per row.
// ---------------------------------------------------------------------------
__global__ __launch_bounds__(128)
void final_kernel(const float* __restrict__ ip, const float* __restrict__ disc,
                  const float* __restrict__ pw, const float* __restrict__ pb,
                  float* __restrict__ out) {
    __shared__ float red[256];
    const int i = blockIdx.x, tid = threadIdx.x;
    const float* row = ip + (size_t)i * D2H;
    float a0 = 0.f, a1 = 0.f;
    for (int d = tid; d < D2H; d += 128) {
        float bef;
        if (i == 0) bef = 0.0f;
        else { float v = row[d - D2H]; if (i >= 2) v = fmaxf(v, row[d - 2 * D2H]); bef = v; }
        float aft;
        if (i == NSENT - 1) aft = 0.0f;
        else { float v = row[d + D2H]; if (i <= NSENT - 3) v = fmaxf(v, row[d + 2 * D2H]); aft = v; }
        float cur = row[d];
        a0 += bef * pw[d]        + aft * pw[D2H + d]        + cur * pw[2 * D2H + d];
        a1 += bef * pw[3081 + d] + aft * pw[3081 + D2H + d] + cur * pw[3081 + 2 * D2H + d];
    }
    if (tid < 9) {
        float dv = disc[(size_t)i * 9 + tid];
        a0 += dv * pw[3072 + tid];
        a1 += dv * pw[3081 + 3072 + tid];
    }
    red[tid] = a0; red[128 + tid] = a1; __syncthreads();
    for (int st = 64; st > 0; st >>= 1) {
        if (tid < st) { red[tid] += red[tid + st]; red[128 + tid] += red[128 + tid + st]; }
        __syncthreads();
    }
    if (tid == 0) { out[i * 2 + 0] = red[0] + pb[0]; out[i * 2 + 1] = red[128] + pb[1]; }
}

// ---------------------------------------------------------------------------
// Host orchestration
// ---------------------------------------------------------------------------
extern "C" void kernel_launch(void* const* d_in, const int* in_sizes, int n_in,
                              void* d_out, int out_size, void* d_ws, size_t ws_size,
                              hipStream_t stream) {
    (void)in_sizes; (void)n_in; (void)out_size; (void)ws_size;

    const float* sentence = (const float*)d_in[0];
    const float* disc     = (const float*)d_in[1];
    const float* cW_ih_f  = (const float*)d_in[2];
    const float* cb_f     = (const float*)d_in[4];
    const float* cW_ih_b  = (const float*)d_in[5];
    const float* cb_b     = (const float*)d_in[7];
    const float* dW_ih_f  = (const float*)d_in[8];
    const float* dW_hh_f  = (const float*)d_in[9];
    const float* db_f     = (const float*)d_in[10];
    const float* dW_ih_b  = (const float*)d_in[11];
    const float* dW_hh_b  = (const float*)d_in[12];
    const float* db_b     = (const float*)d_in[13];
    const float* ws1_w    = (const float*)d_in[14];
    const float* ws1_b    = (const float*)d_in[15];
    const float* ws2_w    = (const float*)d_in[16];
    const float* ws2_b    = (const float*)d_in[17];
    const float* ip_w     = (const float*)d_in[18];
    const float* ip_b     = (const float*)d_in[19];
    const float* pred_w   = (const float*)d_in[20];
    const float* pred_b   = (const float*)d_in[21];
    float* out = (float*)d_out;

    // bump allocator over d_ws
    char* base = (char*)d_ws;
    size_t off = 0;
    auto alloc = [&](size_t bytes) -> char* {
        char* p = base + off;
        off = (off + bytes + 255) & ~(size_t)255;
        return p;
    };
    bf16*  sentA  = (bf16*)alloc((size_t)NSENT * EDIM * 2);
    bf16*  Wcf    = (bf16*)alloc((size_t)G4H * EDIM * 2);
    bf16*  Wcb    = (bf16*)alloc((size_t)G4H * EDIM * 2);
    bf16*  Ws1    = (bf16*)alloc((size_t)D2H * D2H * 2);
    bf16*  Wip    = (bf16*)alloc((size_t)EDIM * D2H * 2);
    bf16*  Wdf    = (bf16*)alloc((size_t)G4H * EDIM * 2);
    bf16*  Wdb    = (bf16*)alloc((size_t)G4H * EDIM * 2);
    float* gates  = (float*)alloc((size_t)NSENT * G4H * 4);
    float* o2f    = (float*)alloc((size_t)NSENT * D2H * 4);
    bf16*  o2b    = (bf16*)alloc((size_t)NSENT * D2H * 2);
    float* att    = (float*)alloc((size_t)NSENT * D2H * 4);
    float* svec   = (float*)alloc((size_t)NSENT * 4);
    float* scl    = (float*)alloc(256);
    bf16*  sente  = (bf16*)alloc((size_t)NSENT * D2H * 2);
    bf16*  innerb = (bf16*)alloc((size_t)NSENT * EDIM * 2);
    float* Xf     = (float*)alloc((size_t)NSENT * G4H * 4);
    float* Xb     = (float*)alloc((size_t)NSENT * G4H * 4);
    float* ipred  = (float*)alloc((size_t)NSENT * D2H * 4);
    float* hbuf   = (float*)alloc(2 * 2 * HDIM * 4);
    unsigned int* ctr = (unsigned int*)alloc(256);

    auto cvt = [&](const float* src, bf16* dst, int n) {
        cvt_bf16_kernel<<<(n + 255) / 256, 256, 0, stream>>>(src, dst, n);
    };
    cvt(sentence, sentA, NSENT * EDIM);
    cvt(cW_ih_f, Wcf, G4H * EDIM);
    cvt(cW_ih_b, Wcb, G4H * EDIM);
    cvt(ws1_w,   Ws1, D2H * D2H);
    cvt(ip_w,    Wip, EDIM * D2H);
    cvt(dW_ih_f, Wdf, G4H * EDIM);
    cvt(dW_ih_b, Wdb, G4H * EDIM);

    const dim3 blk(256);
    // Context encoder: gates = sentence @ W_ih^T + b, then cell epilogue
    wmma_gemm_kernel<EPI_RAW><<<dim3(G4H / 64, NSENT / 128), blk, 0, stream>>>(
        sentA, Wcf, cb_f, gates, nullptr, NSENT, G4H, EDIM);
    hgate_kernel<<<(NSENT * HDIM + 255) / 256, blk, 0, stream>>>(gates, o2f, o2b, 0);
    wmma_gemm_kernel<EPI_RAW><<<dim3(G4H / 64, NSENT / 128), blk, 0, stream>>>(
        sentA, Wcb, cb_b, gates, nullptr, NSENT, G4H, EDIM);
    hgate_kernel<<<(NSENT * HDIM + 255) / 256, blk, 0, stream>>>(gates, o2f, o2b, HDIM);

    // Attention: att = tanh(outp2 @ ws1^T + b); s = att @ ws2^T; scale = sum softmax(s)
    wmma_gemm_kernel<EPI_TANH_F32><<<dim3(D2H / 64, NSENT / 128), blk, 0, stream>>>(
        o2b, Ws1, ws1_b, att, nullptr, NSENT, D2H, D2H);
    smv_kernel<<<NSENT / 256, blk, 0, stream>>>(att, ws2_w, ws2_b, svec);
    scale_kernel<<<1, 1024, 0, stream>>>(svec, scl);
    scale_sent_kernel<<<(NSENT * D2H + 255) / 256, blk, 0, stream>>>(o2f, scl, sente);

    // _inner = tanh(sent_enc @ ip_w^T + b)  (bf16 out, feeds next GEMMs)
    wmma_gemm_kernel<EPI_TANH_BF16><<<dim3(EDIM / 64, NSENT / 128), blk, 0, stream>>>(
        sente, Wip, ip_b, nullptr, innerb, NSENT, EDIM, D2H);

    // Discourse input gates for both directions (hoisted out of the scan)
    wmma_gemm_kernel<EPI_RAW><<<dim3(G4H / 64, NSENT / 128), blk, 0, stream>>>(
        innerb, Wdf, db_f, Xf, nullptr, NSENT, G4H, EDIM);
    wmma_gemm_kernel<EPI_RAW><<<dim3(G4H / 64, NSENT / 128), blk, 0, stream>>>(
        innerb, Wdb, db_b, Xb, nullptr, NSENT, G4H, EDIM);

    // Serial bidirectional scan: LDS-resident W_hh, 32 persistent workgroups
    rec_init_kernel<<<(2 * 2 * HDIM + 255) / 256, blk, 0, stream>>>(hbuf, ctr);
    (void)hipFuncSetAttribute((const void*)lstm_rec_kernel,
                              hipFuncAttributeMaxDynamicSharedMemorySize, REC_LDS_BYTES);
    lstm_rec_kernel<<<32, 512, REC_LDS_BYTES, stream>>>(Xf, Xb, dW_hh_f, dW_hh_b, hbuf, ctr, ipred);

    // Window max-pool + concat + final linear
    final_kernel<<<NSENT, 128, 0, stream>>>(ipred, disc, pred_w, pred_b, out);
}
